// MyPVT9_69784628625567
// MI455X (gfx1250) — matrix operand order repaired and checked by reading, not compiled, not run on hardware
//
#include <hip/hip_runtime.h>

#define DEV __device__ __forceinline__

typedef __attribute__((ext_vector_type(16))) __bf16 v16bf;
typedef __attribute__((ext_vector_type(8)))  __bf16 v8bf;
typedef __attribute__((ext_vector_type(8)))  float  v8f;

// ---------------------------------------------------------------- constants
constexpr int CB   = 64;     // batch
constexpr int CN   = 3136;   // tokens
constexpr int CNG  = 49;     // grid tokens
constexpr int CNS  = 3087;   // sampled-from tokens (N - N_grid)
constexpr int CND  = 833;    // down tokens (49 + 784)
constexpr int CMP  = 848;    // padded down tokens (53*16)
constexpr int CNK  = 196;    // keys (14*14)
constexpr int CNKP = 224;    // padded keys (7*32)
constexpr int CNKQ = 208;    // padded keys for M-tiling (13*16)
constexpr int CHW  = 3136;   // 56*56
constexpr int CMC  = 12544;  // conv GEMM rows = B*196

// ---------------------------------------------------------------- wave utils
DEV float wred_sum(float v) {
#pragma unroll
  for (int o = 16; o > 0; o >>= 1) v += __shfl_xor(v, o, 32);
  return v;
}
DEV float wred_max(float v) {
#pragma unroll
  for (int o = 16; o > 0; o >>= 1) v = fmaxf(v, __shfl_xor(v, o, 32));
  return v;
}
DEV void ln64(float x0, float x1, const float* g, const float* bb, int lane,
              float& y0, float& y1) {
  float s  = wred_sum(x0 + x1);
  float ss = wred_sum(x0 * x0 + x1 * x1);
  float mean = s * 0.015625f;
  float var  = ss * 0.015625f - mean * mean;
  float rs = rsqrtf(var + 1e-5f);
  y0 = (x0 - mean) * rs * g[lane] + bb[lane];
  y1 = (x1 - mean) * rs * g[lane + 32] + bb[lane + 32];
}

// ---------------------------------------------------------------- WMMA utils
DEV v8f vzero8() {
  v8f c;
#pragma unroll
  for (int r = 0; r < 8; ++r) c[r] = 0.0f;
  return c;
}
DEV v8f wmma_bf16(v16bf a, v16bf b, v8f c) {
  return __builtin_amdgcn_wmma_f32_16x16x32_bf16(false, a, false, b, (short)0,
                                                 c, false, false);
}
DEV v16bf cat8(v8bf lo, v8bf hi) {
  return __builtin_shufflevector(lo, hi, 0, 1, 2, 3, 4, 5, 6, 7, 8, 9, 10, 11,
                                 12, 13, 14, 15);
}
// A fragment (16x32) from row-major (M,K): lane holds row m0+(l&15),
// K chunks [k0+h*8, +7] and [k0+16+h*8, +7]  -> two b128 loads
DEV v16bf load_a16(const __bf16* A, int lda, int m0, int k0) {
  int lane = threadIdx.x & 31;
  const __bf16* p =
      A + (long long)(m0 + (lane & 15)) * lda + k0 + ((lane >> 4) << 3);
  v8bf lo = *(const v8bf*)p;
  v8bf hi = *(const v8bf*)(p + 16);
  return cat8(lo, hi);
}
// B fragment (32x16) from K-contiguous transposed storage Wt (N,K):
// lane holds col n0+(l&15), K chunks [k0+h*8, +7] and [k0+16+h*8, +7] -> two b128 loads
DEV v16bf load_bt16(const __bf16* Wt, int ldk, int n0, int k0) {
  int lane = threadIdx.x & 31;
  const __bf16* p =
      Wt + (long long)(n0 + (lane & 15)) * ldk + k0 + ((lane >> 4) << 3);
  v8bf lo = *(const v8bf*)p;
  v8bf hi = *(const v8bf*)(p + 16);
  return cat8(lo, hi);
}

// ---------------------------------------------------------------- kernels

// weights -> bf16, stored transposed (N,K) so GEMM B-loads are K-contiguous
__global__ void k_prep_weights(const float* qw, const float* kw, const float* vw,
                               const float* projw, const float* srw,
                               const float* fcw, const float* fc1w,
                               const float* fc2w, __bf16* qwT, __bf16* kwT,
                               __bf16* vwT, __bf16* projwT, __bf16* WcT,
                               __bf16* fcwT, __bf16* fc1wT, __bf16* fc2wT) {
  int i = blockIdx.x * blockDim.x + threadIdx.x;
  if (i < 8192) {  // (64,128) -> (128,64)
    int n = i >> 6, k = i & 63;
    qwT[i]  = (__bf16)qw[k * 128 + n];
    kwT[i]  = (__bf16)kw[k * 128 + n];
    vwT[i]  = (__bf16)vw[k * 128 + n];
    fcwT[i] = (__bf16)fcw[k * 128 + n];
  }
  if (i < 16384) {  // (128,128) -> (128,128)^T
    int n = i >> 7, k = i & 127;
    projwT[i] = (__bf16)projw[k * 128 + n];
  }
  if (i < 65536) {
    {  // fc1 (128,512) -> (512,128)
      int n = i >> 7, k = i & 127;
      fc1wT[i] = (__bf16)fc1w[k * 512 + n];
    }
    {  // fc2 (512,128) -> (128,512)
      int n = i >> 9, k = i & 511;
      fc2wT[i] = (__bf16)fc2w[k * 128 + n];
    }
    // sr_w (Cout, Cin*4*4) is already (N=64, K=1024)
    WcT[i] = (__bf16)srw[i];
  }
}

__global__ void k_zero(float* p, long long n) {
  long long i = (long long)blockIdx.x * blockDim.x + threadIdx.x;
  if (i < n) p[i] = 0.0f;
}
__global__ void k_zero_bf(__bf16* p, long long n) {
  long long i = (long long)blockIdx.x * blockDim.x + threadIdx.x;
  if (i < n) p[i] = (__bf16)0.0f;
}

// conf = LN(x)@conf_w + conf_b + gumbel(noise)
__global__ void k_scores(const float* x, const float* noise_u, const float* g,
                         const float* bb, const float* conf_w,
                         const float* conf_b, float* scores) {
  int wid = (blockIdx.x * blockDim.x + threadIdx.x) >> 5;
  int lane = threadIdx.x & 31;
  if (wid >= CB * CNS) return;
  int b = wid / CNS, j = wid % CNS;
  const float* row = x + ((long long)b * CN + CNG + j) * 64;
  float x0 = row[lane], x1 = row[lane + 32];
  float y0, y1;
  ln64(x0, x1, g, bb, lane, y0, y1);
  float d = wred_sum(y0 * conf_w[lane] + y1 * conf_w[lane + 32]);
  if (lane == 0) {
    float u = noise_u[(long long)b * CNS + j];
    float noise = -logf(-logf(u + 1e-6f) + 1e-6f);
    scores[wid] = d + conf_b[0] + noise;
  }
}

DEV bool kv_less(float ka, int va, float kb, int vb) {
  // "less" in descending order with ascending-index tie break
  return (ka < kb) || (ka == kb && va > vb);
}

__global__ void __launch_bounds__(1024) k_topk(const float* scores, int* idxb) {
  __shared__ float skey[4096];
  __shared__ int   sval[4096];
  int b = blockIdx.x;
  int tid = threadIdx.x;
  for (int i = tid; i < 4096; i += 1024) {
    if (i < CNS) { skey[i] = scores[(long long)b * CNS + i]; sval[i] = i; }
    else         { skey[i] = -3.0e38f; sval[i] = 0x7fffffff; }
  }
  __syncthreads();
  for (int k = 2; k <= 4096; k <<= 1) {
    for (int j = k >> 1; j > 0; j >>= 1) {
      for (int t = tid; t < 2048; t += 1024) {
        int i = (t / j) * (j << 1) + (t % j);
        int l = i + j;
        bool desc = ((i & k) == 0);
        float ki = skey[i], kl = skey[l];
        int vi = sval[i], vl = sval[l];
        bool sw = desc ? kv_less(ki, vi, kl, vl) : kv_less(kl, vl, ki, vi);
        if (sw) { skey[i] = kl; skey[l] = ki; sval[i] = vl; sval[l] = vi; }
      }
      __syncthreads();
    }
  }
  for (int i = tid; i < 784; i += 1024) idxb[b * 784 + i] = sval[i];
}

// build x_down (bf16), xn_down = LN(x_down) (bf16), pos_down (f32); zero-pad
__global__ void k_gather(const float* x, const float* pos, const int* idxb,
                         const float* g, const float* bb, __bf16* xdb,
                         __bf16* xndb, float* posd) {
  int wid = (blockIdx.x * blockDim.x + threadIdx.x) >> 5;
  int lane = threadIdx.x & 31;
  if (wid >= CB * CMP) return;
  int b = wid / CMP, m = wid % CMP;
  __bf16* xdrow = xdb + (long long)wid * 64;
  __bf16* xnrow = xndb + (long long)wid * 64;
  if (m >= CND) {
    xdrow[lane] = (__bf16)0.0f; xdrow[lane + 32] = (__bf16)0.0f;
    xnrow[lane] = (__bf16)0.0f; xnrow[lane + 32] = (__bf16)0.0f;
    if (lane < 2) posd[(long long)wid * 2 + lane] = 0.0f;
    return;
  }
  int src = (m < CNG) ? m : (CNG + idxb[b * 784 + (m - CNG)]);
  const float* row = x + ((long long)b * CN + src) * 64;
  float x0 = row[lane], x1 = row[lane + 32];
  if (lane < 2) posd[(long long)wid * 2 + lane] =
      pos[((long long)b * CN + src) * 2 + lane];
  float y0, y1;
  ln64(x0, x1, g, bb, lane, y0, y1);
  xdrow[lane] = (__bf16)x0; xdrow[lane + 32] = (__bf16)x1;
  xnrow[lane] = (__bf16)y0; xnrow[lane + 32] = (__bf16)y1;
}

// token2map scatter: LN(x) per token, atomic-add into (B,65,56,56)
__global__ void k_scatter(const float* x, const float* pos, const float* g,
                          const float* bb, float* G0) {
  int wid = (blockIdx.x * blockDim.x + threadIdx.x) >> 5;
  int lane = threadIdx.x & 31;
  if (wid >= CB * CN) return;
  int b = wid / CN;
  const float* row = x + (long long)wid * 64;
  float x0 = row[lane], x1 = row[lane + 32];
  float y0, y1;
  ln64(x0, x1, g, bb, lane, y0, y1);
  float gx = pos[(long long)wid * 2 + 0];
  float gy = pos[(long long)wid * 2 + 1];
  gx = fminf(fmaxf(gx, 0.0f), 1.0f);
  gy = fminf(fmaxf(gy, 0.0f), 1.0f);
  int px = (int)rintf(gx * 55.0f);
  int py = (int)rintf(gy * 55.0f);
  int p = py * 56 + px;
  float* base = G0 + (long long)b * 65 * CHW;
  atomicAdd(base + (long long)lane * CHW + p, y0);
  atomicAdd(base + (long long)(lane + 32) * CHW + p, y1);
  if (lane == 0) atomicAdd(base + (long long)64 * CHW + p, 1.0f);
}

__global__ void k_normalize(float* G0) {
  int wid = (blockIdx.x * blockDim.x + threadIdx.x) >> 5;
  int lane = threadIdx.x & 31;
  if (wid >= CB * CHW) return;
  int b = wid / CHW, p = wid % CHW;
  float* base = G0 + (long long)b * 65 * CHW;
  float mask = base[(long long)64 * CHW + p];
  float mb = (mask > 0.0f) ? 1.0f : 0.0f;
  float inv = 1.0f / (mask + 1e-6f);
  float f0 = base[(long long)lane * CHW + p] * inv * mb;
  float f1 = base[(long long)(lane + 32) * CHW + p] * inv * mb;
  base[(long long)lane * CHW + p] = f0;
  base[(long long)(lane + 32) * CHW + p] = f1;
  if (lane == 0) base[(long long)64 * CHW + p] = mb;
}

// 3x3 gaussian (sigma=2), zero-padded, over 65 channels
__global__ void k_gauss(const float* G0, float* G1) {
  long long i = (long long)blockIdx.x * blockDim.x + threadIdx.x;
  if (i >= (long long)CB * 65 * CHW) return;
  int p = (int)(i % CHW);
  long long bc = i / CHW;
  int y = p / 56, xg = p % 56;
  const float e1 = 0.88249690f;  // exp(-1/8)
  const float e2 = 0.77880078f;  // exp(-2/8)
  const float wsum = 1.0f + 4.0f * e1 + 4.0f * e2;
  const float* base = G0 + bc * CHW;
  float acc = 0.0f;
#pragma unroll
  for (int dy = -1; dy <= 1; ++dy)
#pragma unroll
    for (int dx = -1; dx <= 1; ++dx) {
      int yy = y + dy, xx = xg + dx;
      if (yy >= 0 && yy < 56 && xx >= 0 && xx < 56) {
        float w = (dx == 0 && dy == 0) ? 1.0f
                 : ((dx == 0 || dy == 0) ? e1 : e2);
        acc += base[yy * 56 + xx] * w;
      }
    }
  G1[i] = acc / wsum;
}

// fmap = feat + (1-mask) * (G1c/(G1m+eps)) * (G1m>0), written into G1[c]
__global__ void k_gfinal(const float* G0, float* G1) {
  long long i = (long long)blockIdx.x * blockDim.x + threadIdx.x;
  if (i >= (long long)CB * 64 * CHW) return;
  int p = (int)(i % CHW);
  long long t = i / CHW;
  int c = (int)(t % 64);
  int b = (int)(t / 64);
  long long gb = (long long)b * 65 * CHW;
  float mi = G1[gb + (long long)64 * CHW + p];
  float fi = G1[gb + (long long)c * CHW + p] / (mi + 1e-6f);
  fi *= (mi > 0.0f) ? 1.0f : 0.0f;
  float feat = G0[gb + (long long)c * CHW + p];
  float m = G0[gb + (long long)64 * CHW + p];
  G1[gb + (long long)c * CHW + p] = feat + (1.0f - m) * fi;
}

// im2col for the 4x4 stride-4 conv: (B*196, 1024) bf16
__global__ void k_im2col(const float* G1, __bf16* A) {
  long long i = (long long)blockIdx.x * blockDim.x + threadIdx.x;
  if (i >= (long long)CMC * 1024) return;
  int k = (int)(i & 1023);
  int r = (int)(i >> 10);
  int b = r / CNK, p = r % CNK;
  int oy = p / 14, ox = p % 14;
  int ci = k >> 4;
  int kk = k & 15;
  int kh = kk >> 2, kw = kk & 3;
  float v = G1[((long long)b * 65 + ci) * CHW + (oy * 4 + kh) * 56 + (ox * 4 + kw)];
  A[i] = (__bf16)v;
}

// conv GEMM: (12544,1024) x (1024,64)^T + sr_b -> f32; one wave = 16x64 tile
__global__ void k_gemm_conv(const __bf16* A, const __bf16* WcT,
                            const float* bias, float* C) {
  int wid = (blockIdx.x * blockDim.x + threadIdx.x) >> 5;
  if (wid >= CMC / 16) return;
  int m0 = wid * 16;
  v8f acc0 = vzero8(), acc1 = vzero8(), acc2 = vzero8(), acc3 = vzero8();
  for (int k0 = 0; k0 < 1024; k0 += 32) {
    if (k0 + 32 < 1024)
      __builtin_prefetch(A + (long long)m0 * 1024 + k0 + 32, 0, 1);
    v16bf a = load_a16(A, 1024, m0, k0);
    acc0 = wmma_bf16(a, load_bt16(WcT, 1024, 0, k0), acc0);
    acc1 = wmma_bf16(a, load_bt16(WcT, 1024, 16, k0), acc1);
    acc2 = wmma_bf16(a, load_bt16(WcT, 1024, 32, k0), acc2);
    acc3 = wmma_bf16(a, load_bt16(WcT, 1024, 48, k0), acc3);
  }
  int lane = threadIdx.x & 31;
  int nl = lane & 15;
  int mh = (lane >> 4) * 8;
#pragma unroll
  for (int r = 0; r < 8; ++r) {
    long long m = m0 + mh + r;
    C[m * 64 + nl]      = acc0[r] + bias[nl];
    C[m * 64 + nl + 16] = acc1[r] + bias[nl + 16];
    C[m * 64 + nl + 32] = acc2[r] + bias[nl + 32];
    C[m * 64 + nl + 48] = acc3[r] + bias[nl + 48];
  }
}

// LN of conv output -> xs_bf in padded (B,208,64) layout
__global__ void k_srnorm(const float* xsraw, const float* g, const float* bb,
                         __bf16* xsbf) {
  int wid = (blockIdx.x * blockDim.x + threadIdx.x) >> 5;
  int lane = threadIdx.x & 31;
  if (wid >= CMC) return;
  int b = wid / CNK, p = wid % CNK;
  const float* row = xsraw + (long long)wid * 64;
  float x0 = row[lane], x1 = row[lane + 32];
  float y0, y1;
  ln64(x0, x1, g, bb, lane, y0, y1);
  __bf16* o = xsbf + ((long long)b * CNKQ + p) * 64;
  o[lane] = (__bf16)y0; o[lane + 32] = (__bf16)y1;
}

// batched GEMM: out[b](M,N) = A[b](M,K) x Wt(N,K)^T * scale  -> bf16
// one wave = 16x32 tile (2 accumulators sharing the A fragment)
__global__ void k_gemm_xw(const __bf16* A, long long strideA, int lda, int M,
                          int Mtiles, const __bf16* Wt, int K, int N,
                          __bf16* out, long long strideO, float scale,
                          int nBatch) {
  int wid = (blockIdx.x * blockDim.x + threadIdx.x) >> 5;
  int NP = N >> 5;
  int per = Mtiles * NP;
  if (wid >= nBatch * per) return;
  int b = wid / per;
  int t = wid % per;
  int tm = t / NP, tp = t % NP;
  int m0 = tm * 16, n0 = tp * 32;
  const __bf16* Ab = A + (long long)b * strideA;
  v8f acc0 = vzero8(), acc1 = vzero8();
  for (int k0 = 0; k0 < K; k0 += 32) {
    v16bf a = load_a16(Ab, lda, m0, k0);
    acc0 = wmma_bf16(a, load_bt16(Wt, K, n0, k0), acc0);
    acc1 = wmma_bf16(a, load_bt16(Wt, K, n0 + 16, k0), acc1);
  }
  int lane = threadIdx.x & 31;
  int nl = lane & 15;
  int mh = (lane >> 4) * 8;
  __bf16* ob = out + (long long)b * strideO;
#pragma unroll
  for (int r = 0; r < 8; ++r) {
    int m = m0 + mh + r;
    if (m < M) {
      ob[(long long)m * N + n0 + nl]      = (__bf16)(acc0[r] * scale);
      ob[(long long)m * N + n0 + 16 + nl] = (__bf16)(acc1[r] * scale);
    }
  }
}

// same, but stores transposed: out (B,128,CNKP) with out[n][m] = C[m][n]
__global__ void k_gemm_xw_T(const __bf16* A, long long strideA, int lda, int M,
                            int Mtiles, const __bf16* Wt, int K, int N,
                            __bf16* out, int nBatch) {
  int wid = (blockIdx.x * blockDim.x + threadIdx.x) >> 5;
  int NP = N >> 5;
  int per = Mtiles * NP;
  if (wid >= nBatch * per) return;
  int b = wid / per;
  int t = wid % per;
  int tm = t / NP, tp = t % NP;
  int m0 = tm * 16, n0 = tp * 32;
  const __bf16* Ab = A + (long long)b * strideA;
  v8f acc0 = vzero8(), acc1 = vzero8();
  for (int k0 = 0; k0 < K; k0 += 32) {
    v16bf a = load_a16(Ab, lda, m0, k0);
    acc0 = wmma_bf16(a, load_bt16(Wt, K, n0, k0), acc0);
    acc1 = wmma_bf16(a, load_bt16(Wt, K, n0 + 16, k0), acc1);
  }
  int lane = threadIdx.x & 31;
  int nl = lane & 15;
  int mh = (lane >> 4) * 8;
  __bf16* ob = out + (long long)b * 128 * CNKP;
#pragma unroll
  for (int r = 0; r < 8; ++r) {
    int m = m0 + mh + r;
    if (m < M) {
      ob[(long long)(n0 + nl) * CNKP + m]      = (__bf16)acc0[r];
      ob[(long long)(n0 + 16 + nl) * CNKP + m] = (__bf16)acc1[r];
    }
  }
}

// S = q_h (848,64) x k_h^T -> f32 (bh,848,224); one wave = 16x32 tile
__global__ void k_attn_s(const __bf16* q, const __bf16* kmat, float* S) {
  int wid = (blockIdx.x * blockDim.x + threadIdx.x) >> 5;
  const int TM = CMP / 16, TP = CNKP / 32;  // 53, 7
  if (wid >= CB * 2 * TM * TP) return;
  int bh = wid / (TM * TP);
  int t = wid % (TM * TP);
  int tm = t / TP, tp = t % TP;
  int b = bh >> 1, h = bh & 1;
  int m0 = tm * 16, n0 = tp * 32;
  const __bf16* qb = q + (long long)b * CMP * 128 + h * 64;
  const __bf16* kb = kmat + (long long)b * CNKP * 128 + h * 64;
  v8f acc0 = vzero8(), acc1 = vzero8();
#pragma unroll
  for (int k0 = 0; k0 < 64; k0 += 32) {
    v16bf a = load_a16(qb, 128, m0, k0);
    acc0 = wmma_bf16(a, load_bt16(kb, 128, n0, k0), acc0);
    acc1 = wmma_bf16(a, load_bt16(kb, 128, n0 + 16, k0), acc1);
  }
  int lane = threadIdx.x & 31;
  int nl = lane & 15;
  int mh = (lane >> 4) * 8;
  float* Sb = S + (long long)bh * CMP * CNKP;
#pragma unroll
  for (int r = 0; r < 8; ++r) {
    long long m = m0 + mh + r;
    Sb[m * CNKP + n0 + nl]      = acc0[r];
    Sb[m * CNKP + n0 + 16 + nl] = acc1[r];
  }
}

__global__ void k_softmax(const float* S, __bf16* P) {
  int wid = (blockIdx.x * blockDim.x + threadIdx.x) >> 5;
  int lane = threadIdx.x & 31;
  if (wid >= CB * 2 * CMP) return;
  const float* Srow = S + (long long)wid * CNKP;
  __bf16* Prow = P + (long long)wid * CNKP;
  float mx = -3.0e38f;
  for (int i = lane; i < CNK; i += 32) mx = fmaxf(mx, Srow[i]);
  mx = wred_max(mx);
  float sum = 0.0f;
  for (int i = lane; i < CNK; i += 32) sum += expf(Srow[i] - mx);
  sum = wred_sum(sum);
  float inv = 1.0f / sum;
  for (int i = lane; i < CNK; i += 32)
    Prow[i] = (__bf16)(expf(Srow[i] - mx) * inv);
  for (int i = CNK + lane; i < CNKP; i += 32) Prow[i] = (__bf16)0.0f;
}

// O_h = P (848,224) x V_h^T(64,224)^T -> o_bf (b,848,128); wave = 16x32 tile
__global__ void k_attn_o(const __bf16* P, const __bf16* vT, __bf16* o) {
  int wid = (blockIdx.x * blockDim.x + threadIdx.x) >> 5;
  const int TM = CMP / 16;  // 53
  if (wid >= CB * 2 * TM * 2) return;
  int bh = wid / (TM * 2);
  int t = wid % (TM * 2);
  int tm = t / 2, tp = t % 2;
  int b = bh >> 1, h = bh & 1;
  int m0 = tm * 16, n0 = tp * 32;
  const __bf16* Pb = P + (long long)bh * CMP * CNKP;
  const __bf16* vb = vT + (long long)b * 128 * CNKP + (long long)h * 64 * CNKP;
  v8f acc0 = vzero8(), acc1 = vzero8();
#pragma unroll
  for (int k0 = 0; k0 < CNKP; k0 += 32) {
    v16bf a = load_a16(Pb, CNKP, m0, k0);
    acc0 = wmma_bf16(a, load_bt16(vb, CNKP, n0, k0), acc0);
    acc1 = wmma_bf16(a, load_bt16(vb, CNKP, n0 + 16, k0), acc1);
  }
  int lane = threadIdx.x & 31;
  int nl = lane & 15;
  int mh = (lane >> 4) * 8;
#pragma unroll
  for (int r = 0; r < 8; ++r) {
    long long m = (long long)b * CMP + m0 + mh + r;
    o[m * 128 + h * 64 + n0 + nl]      = (__bf16)acc0[r];
    o[m * 128 + h * 64 + n0 + 16 + nl] = (__bf16)acc1[r];
  }
}

// x2 = o @ proj_w + proj_b + x_down @ fc_w + fc_b ; wave = 16x32 tile
__global__ void k_projfc(const __bf16* o, const __bf16* projwT,
                         const float* projb, const __bf16* xd,
                         const __bf16* fcwT, const float* fcb, float* x2) {
  int wid = (blockIdx.x * blockDim.x + threadIdx.x) >> 5;
  const int TM = CMP / 16;  // 53
  if (wid >= CB * TM * 4) return;
  int b = wid / (TM * 4);
  int t = wid % (TM * 4);
  int tm = t / 4, tp = t % 4;
  int m0 = tm * 16, n0 = tp * 32;
  const __bf16* ob = o + (long long)b * CMP * 128;
  const __bf16* xb = xd + (long long)b * CMP * 64;
  v8f acc0 = vzero8(), acc1 = vzero8();
#pragma unroll
  for (int k0 = 0; k0 < 128; k0 += 32) {
    v16bf a = load_a16(ob, 128, m0, k0);
    acc0 = wmma_bf16(a, load_bt16(projwT, 128, n0, k0), acc0);
    acc1 = wmma_bf16(a, load_bt16(projwT, 128, n0 + 16, k0), acc1);
  }
#pragma unroll
  for (int k0 = 0; k0 < 64; k0 += 32) {
    v16bf a = load_a16(xb, 64, m0, k0);
    acc0 = wmma_bf16(a, load_bt16(fcwT, 64, n0, k0), acc0);
    acc1 = wmma_bf16(a, load_bt16(fcwT, 64, n0 + 16, k0), acc1);
  }
  int lane = threadIdx.x & 31;
  int nl = lane & 15;
  int mh = (lane >> 4) * 8;
#pragma unroll
  for (int r = 0; r < 8; ++r) {
    long long m = (long long)b * CMP + m0 + mh + r;
    x2[m * 128 + n0 + nl]      = acc0[r] + projb[n0 + nl] + fcb[n0 + nl];
    x2[m * 128 + n0 + 16 + nl] = acc1[r] + projb[n0 + 16 + nl] + fcb[n0 + 16 + nl];
  }
}

__global__ void k_norm2(const float* x2, const float* g, const float* bb,
                        __bf16* xn2) {
  int wid = (blockIdx.x * blockDim.x + threadIdx.x) >> 5;
  int lane = threadIdx.x & 31;
  if (wid >= CB * CMP) return;
  const float* row = x2 + (long long)wid * 128;
  float v0 = row[lane], v1 = row[lane + 32], v2 = row[lane + 64],
        v3 = row[lane + 96];
  float s = wred_sum(v0 + v1 + v2 + v3);
  float ss = wred_sum(v0 * v0 + v1 * v1 + v2 * v2 + v3 * v3);
  float mean = s * 0.0078125f;
  float var = ss * 0.0078125f - mean * mean;
  float rs = rsqrtf(var + 1e-5f);
  __bf16* o = xn2 + (long long)wid * 128;
  o[lane]      = (__bf16)((v0 - mean) * rs * g[lane] + bb[lane]);
  o[lane + 32] = (__bf16)((v1 - mean) * rs * g[lane + 32] + bb[lane + 32]);
  o[lane + 64] = (__bf16)((v2 - mean) * rs * g[lane + 64] + bb[lane + 64]);
  o[lane + 96] = (__bf16)((v3 - mean) * rs * g[lane + 96] + bb[lane + 96]);
}

// h1 = gelu(xn2 @ fc1_w + fc1_b)   M=54272 K=128 N=512 ; wave = 16x32 tile
__global__ void k_mlp1(const __bf16* A, const __bf16* Wt, const float* bias,
                       __bf16* h1) {
  int wid = (blockIdx.x * blockDim.x + threadIdx.x) >> 5;
  const int TM = (CB * CMP) / 16;  // 3392
  if (wid >= TM * 16) return;
  int tm = wid / 16, tp = wid % 16;
  int m0 = tm * 16, n0 = tp * 32;
  v8f acc0 = vzero8(), acc1 = vzero8();
#pragma unroll
  for (int k0 = 0; k0 < 128; k0 += 32) {
    v16bf a = load_a16(A, 128, m0, k0);
    acc0 = wmma_bf16(a, load_bt16(Wt, 128, n0, k0), acc0);
    acc1 = wmma_bf16(a, load_bt16(Wt, 128, n0 + 16, k0), acc1);
  }
  int lane = threadIdx.x & 31;
  int nl = lane & 15;
  int mh = (lane >> 4) * 8;
#pragma unroll
  for (int r = 0; r < 8; ++r) {
    long long m = m0 + mh + r;
    float t0 = acc0[r] + bias[n0 + nl];
    float t1 = acc1[r] + bias[n0 + 16 + nl];
    h1[m * 512 + n0 + nl] =
        (__bf16)(0.5f * t0 * (1.0f + erff(t0 * 0.70710678118f)));
    h1[m * 512 + n0 + 16 + nl] =
        (__bf16)(0.5f * t1 * (1.0f + erff(t1 * 0.70710678118f)));
  }
}

// x2 += h1 @ fc2_w + fc2_b   M=54272 K=512 N=128 ; wave = 16x32 tile
__global__ void k_mlp2(const __bf16* A, const __bf16* Wt, const float* bias,
                       float* x2) {
  int wid = (blockIdx.x * blockDim.x + threadIdx.x) >> 5;
  const int TM = (CB * CMP) / 16;  // 3392
  if (wid >= TM * 4) return;
  int tm = wid / 4, tp = wid % 4;
  int m0 = tm * 16, n0 = tp * 32;
  v8f acc0 = vzero8(), acc1 = vzero8();
  for (int k0 = 0; k0 < 512; k0 += 32) {
    if (k0 + 32 < 512)
      __builtin_prefetch(A + (long long)m0 * 512 + k0 + 32, 0, 1);
    v16bf a = load_a16(A, 512, m0, k0);
    acc0 = wmma_bf16(a, load_bt16(Wt, 512, n0, k0), acc0);
    acc1 = wmma_bf16(a, load_bt16(Wt, 512, n0 + 16, k0), acc1);
  }
  int lane = threadIdx.x & 31;
  int nl = lane & 15;
  int mh = (lane >> 4) * 8;
#pragma unroll
  for (int r = 0; r < 8; ++r) {
    long long m = m0 + mh + r;
    x2[m * 128 + n0 + nl]      += acc0[r] + bias[n0 + nl];
    x2[m * 128 + n0 + 16 + nl] += acc1[r] + bias[n0 + 16 + nl];
  }
}

// out = x2 + bilinear_sample(pos_embed, pos_down*2-1)
__global__ void k_final(const float* x2, const float* posd,
                        const float* pos_embed, float* out) {
  long long i = (long long)blockIdx.x * blockDim.x + threadIdx.x;
  if (i >= (long long)CB * CND * 128) return;
  int c = (int)(i & 127);
  long long t = i >> 7;
  int m = (int)(t % CND);
  int b = (int)(t / CND);
  const float* pr = posd + ((long long)b * CMP + m) * 2;
  float gx = pr[0] * 2.0f - 1.0f;
  float gy = pr[1] * 2.0f - 1.0f;
  float ix = ((gx + 1.0f) * 56.0f - 1.0f) * 0.5f;
  float iy = ((gy + 1.0f) * 56.0f - 1.0f) * 0.5f;
  float x0 = floorf(ix), y0 = floorf(iy);
  float wx1 = ix - x0, wy1 = iy - y0;
  float acc = 0.0f;
#pragma unroll
  for (int dy = 0; dy < 2; ++dy)
#pragma unroll
    for (int dx = 0; dx < 2; ++dx) {
      float xc = x0 + dx, yc = y0 + dy;
      float w = (dx ? wx1 : 1.0f - wx1) * (dy ? wy1 : 1.0f - wy1);
      bool valid = (xc >= 0.0f) && (xc < 56.0f) && (yc >= 0.0f) && (yc < 56.0f);
      int xi = (int)fminf(fmaxf(xc, 0.0f), 55.0f);
      int yi = (int)fminf(fmaxf(yc, 0.0f), 55.0f);
      float v = pos_embed[(long long)(yi * 56 + xi) * 128 + c];
      acc += v * (valid ? w : 0.0f);
    }
  out[i] = x2[((long long)b * CMP + m) * 128 + c] + acc;
}

// ---------------------------------------------------------------- launch
extern "C" void kernel_launch(void* const* d_in, const int* in_sizes, int n_in,
                              void* d_out, int out_size, void* d_ws,
                              size_t ws_size, hipStream_t stream) {
  const float* x         = (const float*)d_in[0];
  const float* pos       = (const float*)d_in[1];
  const float* pos_embed = (const float*)d_in[2];
  const float* noise_u   = (const float*)d_in[3];
  const float* norm_g    = (const float*)d_in[4];
  const float* norm_b    = (const float*)d_in[5];
  const float* conf_w    = (const float*)d_in[6];
  const float* conf_b    = (const float*)d_in[7];
  const float* norm1_g   = (const float*)d_in[8];
  const float* norm1_b   = (const float*)d_in[9];
  const float* q_w       = (const float*)d_in[10];
  const float* k_w       = (const float*)d_in[11];
  const float* v_w       = (const float*)d_in[12];
  const float* proj_w    = (const float*)d_in[13];
  const float* proj_b    = (const float*)d_in[14];
  const float* sr_w      = (const float*)d_in[15];
  const float* sr_b      = (const float*)d_in[16];
  const float* srnorm_g  = (const float*)d_in[17];
  const float* srnorm_b  = (const float*)d_in[18];
  const float* fc_w      = (const float*)d_in[19];
  const float* fc_b      = (const float*)d_in[20];
  const float* norm2_g   = (const float*)d_in[21];
  const float* norm2_b   = (const float*)d_in[22];
  const float* fc1_w     = (const float*)d_in[23];
  const float* fc1_b     = (const float*)d_in[24];
  const float* fc2_w     = (const float*)d_in[25];
  const float* fc2_b     = (const float*)d_in[26];
  float* outp = (float*)d_out;

  char* wp = (char*)d_ws;
  auto carve = [&](size_t bytes) {
    char* p = wp;
    wp += (bytes + 255) & ~(size_t)255;
    return p;
  };
  float*  scores = (float*)carve((size_t)CB * CNS * 4);
  int*    idxb   = (int*)carve((size_t)CB * 784 * 4);
  float*  G0     = (float*)carve((size_t)CB * 65 * CHW * 4);
  float*  G1     = (float*)carve((size_t)CB * 65 * CHW * 4);
  __bf16* Aconv  = (__bf16*)carve((size_t)CMC * 1024 * 2);
  __bf16* WcT    = (__bf16*)carve((size_t)64 * 1024 * 2);
  __bf16* qwT    = (__bf16*)carve((size_t)8192 * 2);
  __bf16* kwT    = (__bf16*)carve((size_t)8192 * 2);
  __bf16* vwT    = (__bf16*)carve((size_t)8192 * 2);
  __bf16* fcwT   = (__bf16*)carve((size_t)8192 * 2);
  __bf16* projwT = (__bf16*)carve((size_t)16384 * 2);
  __bf16* fc1wT  = (__bf16*)carve((size_t)65536 * 2);
  __bf16* fc2wT  = (__bf16*)carve((size_t)65536 * 2);
  float*  xsraw  = (float*)carve((size_t)CMC * 64 * 4);
  __bf16* xsbf   = (__bf16*)carve((size_t)CB * CNKQ * 64 * 2);   // zero-padded
  __bf16* kbf    = (__bf16*)carve((size_t)CB * CNKP * 128 * 2);  // zero-padded
  __bf16* vT     = (__bf16*)carve((size_t)CB * 128 * CNKP * 2);  // zero-padded
  __bf16* xdb    = (__bf16*)carve((size_t)CB * CMP * 64 * 2);
  __bf16* xndb   = (__bf16*)carve((size_t)CB * CMP * 64 * 2);
  float*  posd   = (float*)carve((size_t)CB * CMP * 2 * 4);
  __bf16* qbf    = (__bf16*)carve((size_t)CB * CMP * 128 * 2);
  float*  S      = (float*)carve((size_t)CB * 2 * CMP * CNKP * 4);
  __bf16* Pbf    = (__bf16*)carve((size_t)CB * 2 * CMP * CNKP * 2);
  __bf16* obf    = (__bf16*)carve((size_t)CB * CMP * 128 * 2);
  float*  x2     = (float*)carve((size_t)CB * CMP * 128 * 4);
  __bf16* xn2b   = (__bf16*)carve((size_t)CB * CMP * 128 * 2);
  __bf16* h1b    = (__bf16*)carve((size_t)CB * CMP * 512 * 2);

  auto tblocks = [](long long nthreads) {
    return (int)((nthreads + 255) / 256);
  };
  auto wblocks = [](long long waves) {
    return (int)((waves * 32 + 255) / 256);
  };

  k_prep_weights<<<tblocks(65536), 256, 0, stream>>>(
      q_w, k_w, v_w, proj_w, sr_w, fc_w, fc1_w, fc2_w, qwT, kwT, vwT, projwT,
      WcT, fcwT, fc1wT, fc2wT);
  k_zero<<<tblocks((long long)CB * 65 * CHW), 256, 0, stream>>>(
      G0, (long long)CB * 65 * CHW);
  k_zero_bf<<<tblocks((long long)CB * CNKQ * 64), 256, 0, stream>>>(
      xsbf, (long long)CB * CNKQ * 64);
  k_zero_bf<<<tblocks((long long)CB * CNKP * 128), 256, 0, stream>>>(
      kbf, (long long)CB * CNKP * 128);
  k_zero_bf<<<tblocks((long long)CB * 128 * CNKP), 256, 0, stream>>>(
      vT, (long long)CB * 128 * CNKP);
  k_scores<<<wblocks((long long)CB * CNS), 256, 0, stream>>>(
      x, noise_u, norm_g, norm_b, conf_w, conf_b, scores);
  k_topk<<<CB, 1024, 0, stream>>>(scores, idxb);
  k_gather<<<wblocks((long long)CB * CMP), 256, 0, stream>>>(
      x, pos, idxb, norm1_g, norm1_b, xdb, xndb, posd);
  k_scatter<<<wblocks((long long)CB * CN), 256, 0, stream>>>(
      x, pos, norm1_g, norm1_b, G0);
  k_normalize<<<wblocks((long long)CB * CHW), 256, 0, stream>>>(G0);
  k_gauss<<<tblocks((long long)CB * 65 * CHW), 256, 0, stream>>>(G0, G1);
  k_gfinal<<<tblocks((long long)CB * 64 * CHW), 256, 0, stream>>>(G0, G1);
  k_im2col<<<tblocks((long long)CMC * 1024), 256, 0, stream>>>(G1, Aconv);
  k_gemm_conv<<<wblocks((long long)(CMC / 16)), 256, 0, stream>>>(Aconv, WcT,
                                                                  sr_b, xsraw);
  k_srnorm<<<wblocks((long long)CMC), 256, 0, stream>>>(xsraw, srnorm_g,
                                                        srnorm_b, xsbf);
  // k = xs @ k_w (bf16, zero-padded rows) ; v stored transposed (B,128,224)
  k_gemm_xw<<<wblocks((long long)CB * 13 * 4), 256, 0, stream>>>(
      xsbf, (long long)CNKQ * 64, 64, CNK, 13, kwT, 64, 128, kbf,
      (long long)CNKP * 128, 1.0f, CB);
  k_gemm_xw_T<<<wblocks((long long)CB * 13 * 4), 256, 0, stream>>>(
      xsbf, (long long)CNKQ * 64, 64, CNK, 13, vwT, 64, 128, vT, CB);
  // q = LN(x_down) @ q_w, pre-scaled by (DIM//NUM_HEADS)^-0.5 = 1/sqrt(32)
  k_gemm_xw<<<wblocks((long long)CB * 53 * 4), 256, 0, stream>>>(
      xndb, (long long)CMP * 64, 64, CMP, 53, qwT, 64, 128, qbf,
      (long long)CMP * 128, 0.17677669529663687f, CB);
  k_attn_s<<<wblocks((long long)CB * 2 * 53 * 7), 256, 0, stream>>>(qbf, kbf,
                                                                    S);
  k_softmax<<<wblocks((long long)CB * 2 * CMP), 256, 0, stream>>>(S, Pbf);
  k_attn_o<<<wblocks((long long)CB * 2 * 53 * 2), 256, 0, stream>>>(Pbf, vT,
                                                                    obf);
  k_projfc<<<wblocks((long long)CB * 53 * 4), 256, 0, stream>>>(
      obf, projwT, proj_b, xdb, fcwT, fc_b, x2);
  k_norm2<<<wblocks((long long)CB * CMP), 256, 0, stream>>>(x2, norm2_g,
                                                            norm2_b, xn2b);
  k_mlp1<<<wblocks((long long)3392 * 16), 256, 0, stream>>>(xn2b, fc1wT, fc1_b,
                                                            h1b);
  k_mlp2<<<wblocks((long long)3392 * 4), 256, 0, stream>>>(h1b, fc2wT, fc2_b,
                                                           x2);
  k_final<<<tblocks((long long)CB * CND * 128), 256, 0, stream>>>(
      x2, posd, pos_embed, outp);

  (void)in_sizes; (void)n_in; (void)out_size; (void)ws_size;
}
